// MutualAttention_2619930051055
// MI455X (gfx1250) — matrix operand order repaired and checked by reading, not compiled, not run on hardware
//
#include <hip/hip_runtime.h>

// ---------------- constants ----------------
#define HWN   65536      // H*W
#define BN    2
#define CN    64
#define HN    256
#define WN    256

typedef __attribute__((ext_vector_type(16))) __bf16 bf16x16;
typedef __attribute__((ext_vector_type(8)))  __bf16 bf16x8;
typedef __attribute__((ext_vector_type(8)))  float  floatx8;
typedef __attribute__((ext_vector_type(4)))  int    i32x4;

// native fp32 -> bf16 (backend emits v_cvt_*bf16_f32)
__device__ __forceinline__ __bf16 to_bf(float f)  { return (__bf16)f; }
__device__ __forceinline__ __bf16 to_bf(__bf16 f) { return f; }

// ---- CDNA5 async global->LDS copy (guarded; falls back to vector copy) ----
#if defined(__has_builtin)
#if __has_builtin(__builtin_amdgcn_global_load_async_to_lds_b128)
#define HAVE_ASYNC_LDS 1
#endif
#endif

__device__ __forceinline__ void copy16_g2l(const __bf16* g, __bf16* l) {
#if HAVE_ASYNC_LDS
  __builtin_amdgcn_global_load_async_to_lds_b128(
      (i32x4*)(void*)g, (i32x4*)(void*)l, 0, 0);
#else
  *(bf16x8*)l = *(const bf16x8*)g;
#endif
}
__device__ __forceinline__ void async_join() {
#if HAVE_ASYNC_LDS
  asm volatile("s_wait_asynccnt 0" ::: "memory");
#endif
}

// 16 contiguous bf16 (two 16B loads) -> fragment
__device__ __forceinline__ bf16x16 frag_b16(const __bf16* p) {
  bf16x8 lo = *(const bf16x8*)(p);
  bf16x8 hi = *(const bf16x8*)(p + 8);
  bf16x16 r;
#pragma unroll
  for (int i = 0; i < 8; ++i) { r[i] = lo[i]; r[8 + i] = hi[i]; }
  return r;
}
// A operand, bf16 source: K = {koff..koff+7, 16+koff..16+koff+7}
__device__ __forceinline__ bf16x16 frag_b16_split(const __bf16* p, int koff) {
  bf16x8 lo = *(const bf16x8*)(p + koff);
  bf16x8 hi = *(const bf16x8*)(p + 16 + koff);
  bf16x16 r;
#pragma unroll
  for (int i = 0; i < 8; ++i) { r[i] = lo[i]; r[8 + i] = hi[i]; }
  return r;
}

__device__ __forceinline__ floatx8 wmma_bf16(bf16x16 a, bf16x16 b, floatx8 c) {
  return __builtin_amdgcn_wmma_f32_16x16x32_bf16(false, a, false, b, (short)0, c, false, false);
}

// ---------------------------------------------------------------------------
// Weight pre-swizzle: src fp32 [M][Cin][ntaps] -> bf16 A-fragments:
// dst[(((mt*ntaps+tap)*nch+ch)*32+lane)*16 + s]
// ---------------------------------------------------------------------------
__global__ void k_pack_weights(const float* __restrict__ src, __bf16* __restrict__ dst,
                               int Mtot, int Cin, int ntaps) {
  int nmt = Mtot >> 4, nch = Cin >> 5;
  long total = (long)nmt * ntaps * nch * 32 * 16;
  for (long e = (long)blockIdx.x * blockDim.x + threadIdx.x; e < total;
       e += (long)gridDim.x * blockDim.x) {
    int s = (int)(e & 15); long r = e >> 4;
    int lane = (int)(r & 31); r >>= 5;
    int ch = (int)(r % nch); r /= nch;
    int tap = (int)(r % ntaps); r /= ntaps;
    int mt = (int)r;
    int m = mt * 16 + (lane & 15);
    int kk = (lane >> 4) * 16 + s;
    int cin = ch * 32 + kk;
    dst[e] = to_bf(src[((long)m * Cin + cin) * ntaps + tap]);
  }
}

// ---------------------------------------------------------------------------
// 64x64 x HW GEMM (1x1 conv). InT: float (channel-major) or bf16 pixel-major.
// Output bf16: channel-major (OUTPIX=0) or pixel-major (OUTPIX=1, ostride/ooff).
// ---------------------------------------------------------------------------
template <typename InT, bool INPIX, bool OUTPIX>
__global__ void __launch_bounds__(256) k_gemm64(
    const InT* __restrict__ in, long in_bs,
    const __bf16* __restrict__ wp, long wp_bs,
    const float* __restrict__ bias,
    __bf16* __restrict__ out, long out_bs, int ostride, int ooff) {
  __shared__ __align__(16) __bf16 tin[128][72];   // [pixel][chan]
  const int b = blockIdx.y;
  const long p0 = (long)blockIdx.x * 128;
  const InT* inb = in + (long)b * in_bs;
  const __bf16* wpb = wp + (long)b * wp_bs;
  const int tid = threadIdx.x;

  if (INPIX) {   // pixel-major bf16: contiguous 16B chunks -> async to LDS
    for (int i = tid; i < 128 * 4; i += 256) {
      int p = i >> 2, slot = i & 3;
      copy16_g2l((const __bf16*)inb + (p0 + p) * CN + slot * 8, &tin[p][slot * 8]);
    }
    async_join();
  } else {       // channel-major fp32: coalesced along pixels, convert
    for (int e = tid; e < 64 * 128; e += 256) {
      int c = e >> 7, p = e & 127;
      tin[p][c] = to_bf(inb[(long)c * HWN + p0 + p]);
    }
    if (p0 + 128 < HWN) {
      int c = tid >> 2, p = (tid & 3) * 32;
      __builtin_prefetch(&inb[(long)c * HWN + p0 + 128 + p], 0, 1);
    }
  }
  __syncthreads();

  const int wave = tid >> 5, lane = tid & 31;
  const int n = lane & 15, khalf = lane >> 4;
  const int n0 = wave * 16;
#pragma unroll
  for (int mt = 0; mt < 4; ++mt) {
    floatx8 acc = {};
#pragma unroll
    for (int ks = 0; ks < 2; ++ks) {
      bf16x16 a = frag_b16(&wpb[(((long)mt * 2 + ks) * 32 + lane) * 16]);
      bf16x16 bb = frag_b16(&tin[n0 + n][ks * 32 + khalf * 16]);
      acc = wmma_bf16(a, bb, acc);
    }
    const int m0 = mt * 16 + ((lane < 16) ? 0 : 8);
    if (OUTPIX) {   // 8 consecutive channels -> one 16B store per lane
      bf16x8 v8;
#pragma unroll
      for (int r = 0; r < 8; ++r) v8[r] = to_bf(acc[r] + bias[m0 + r]);
      *(bf16x8*)&out[(long)b * out_bs + (p0 + n0 + n) * (long)ostride + ooff + m0] = v8;
    } else {
#pragma unroll
      for (int r = 0; r < 8; ++r)
        out[(long)b * out_bs + (long)(m0 + r) * HWN + p0 + n0 + n] = to_bf(acc[r] + bias[m0 + r]);
    }
  }
}

// ---------------------------------------------------------------------------
// per-channel L2 norms of Q, K(x_e), K(x_i) (bf16, channel-major)
// ---------------------------------------------------------------------------
__global__ void __launch_bounds__(256) k_norms(
    const __bf16* __restrict__ Q, const __bf16* __restrict__ Kxe,
    const __bf16* __restrict__ Kxi, float* __restrict__ norms) {
  const int c = blockIdx.x, b = blockIdx.y, t = blockIdx.z;
  const __bf16* src = (t == 0) ? Q : (t == 1) ? Kxe : Kxi;
  const __bf16* p = src + ((long)b * CN + c) * HWN;
  float s = 0.f;
  for (int i = threadIdx.x; i < HWN; i += 256) { float v = (float)p[i]; s += v * v; }
  __shared__ float red[256];
  red[threadIdx.x] = s; __syncthreads();
  for (int off = 128; off > 0; off >>= 1) {
    if (threadIdx.x < off) red[threadIdx.x] += red[threadIdx.x + off];
    __syncthreads();
  }
  if (threadIdx.x == 0) norms[((long)t * BN + b) * CN + c] = sqrtf(red[0]);
}

__global__ void k_zero(float* p, long n) {
  for (long i = (long)blockIdx.x * blockDim.x + threadIdx.x; i < n;
       i += (long)gridDim.x * blockDim.x) p[i] = 0.f;
}

// ---------------------------------------------------------------------------
// Attention scores via long-K WMMA on bf16 Q/K. One 16x16 tile = head PAIR;
// diagonal 8x8 blocks accumulated atomically over hw slices.
// ---------------------------------------------------------------------------
__global__ void __launch_bounds__(256) k_scores(
    const __bf16* __restrict__ Q, const __bf16* __restrict__ Kxe,
    const __bf16* __restrict__ Kxi, float* __restrict__ S) {
  const int b = blockIdx.z;
  const int hp = blockIdx.y & 3, src = blockIdx.y >> 2;
  const __bf16* K = src ? Kxi : Kxe;
  const int tid = threadIdx.x, wave = tid >> 5, lane = tid & 31;
  const int n = lane & 15, khalf = lane >> 4;
  const int koff = (lane < 16) ? 0 : 8;
  const __bf16* qrow = Q + ((long)b * CN + hp * 16 + n) * HWN;
  const __bf16* krow = K + ((long)b * CN + hp * 16 + n) * HWN;
  long base = (long)blockIdx.x * 16384 + (long)wave * 2048;
  floatx8 acc = {};
  for (int it = 0; it < 64; ++it) {
    long pos = base + (long)it * 32;
    bf16x16 a = frag_b16(qrow + pos + khalf * 16);
    bf16x16 bb = frag_b16(krow + pos + khalf * 16);
    acc = wmma_bf16(a, bb, acc);
  }
#pragma unroll
  for (int r = 0; r < 8; ++r) {
    int M = ((lane < 16) ? 0 : 8) + r;
    if ((M < 8) == (n < 8)) {
      int h = hp * 2 + (M >= 8);
      atomicAdd(&S[(((long)b * 2 + src) * 8 + h) * 64 + (M & 7) * 8 + (n & 7)], acc[r]);
    }
  }
}

// ---------------------------------------------------------------------------
// softmax + fold Wo: Meff[b][s] = Wo @ blockdiag(attn[b][src(s)])
// ---------------------------------------------------------------------------
__global__ void __launch_bounds__(256) k_build_meff(
    const float* __restrict__ S, const float* __restrict__ norms,
    const float* __restrict__ temp, const float* __restrict__ wo,
    float* __restrict__ Meff) {
  __shared__ float attn_s[BN][2][8][8][8];
  const int tid = threadIdx.x;
  {
    int b = tid >> 7, src = (tid >> 6) & 1, h = (tid >> 3) & 7, i = tid & 7;
    float nq = fmaxf(norms[(0 * BN + b) * CN + h * 8 + i], 1e-12f);
    float t = temp[h];
    float row[8], mx = -3.0e38f;
#pragma unroll
    for (int j = 0; j < 8; ++j) {
      float nk = fmaxf(norms[((1 + src) * BN + b) * CN + h * 8 + j], 1e-12f);
      float v = S[(((long)b * 2 + src) * 8 + h) * 64 + i * 8 + j] / (nq * nk) * t;
      row[j] = v; mx = fmaxf(mx, v);
    }
    float den = 0.f;
#pragma unroll
    for (int j = 0; j < 8; ++j) { row[j] = expf(row[j] - mx); den += row[j]; }
#pragma unroll
    for (int j = 0; j < 8; ++j) attn_s[b][src][h][i][j] = row[j] / den;
  }
  __syncthreads();
  for (int e = tid; e < BN * 3 * 64 * 64; e += 256) {
    int c = e & 63, o = (e >> 6) & 63;
    int rest = e >> 12, s = rest % 3, b = rest / 3;
    int src = (s == 2) ? 1 : 0, h = c >> 3, j = c & 7;
    float acc = 0.f;
#pragma unroll
    for (int i = 0; i < 8; ++i) acc += wo[o * 64 + h * 8 + i] * attn_s[b][src][h][i][j];
    Meff[(((long)b * 3 + s) * 64 + o) * 64 + c] = acc;
  }
}

// ---------------------------------------------------------------------------
// implicit-GEMM 3x3 conv, reflect padding. Input is bf16 PIXEL-MAJOR [b][hw][CIN]
// so the LDS tile staging is pure contiguous 16B copies (async load-to-LDS).
// NMT=8: dual weight set (w_init||w1) -> out0/out1. Bias dropped (cancels in BN).
// Output fp32 channel-major (feeds BN stats).
// ---------------------------------------------------------------------------
template <int CIN, int NMT>
__global__ void __launch_bounds__(256) k_conv3x3(
    const __bf16* __restrict__ in, const __bf16* __restrict__ wp,
    float* __restrict__ out0, float* __restrict__ out1) {
  constexpr int NCH = CIN / 32;
  constexpr int MPW = NMT / 2;   // m-tiles per wave
  __shared__ __align__(16) __bf16 tin[3][66][32];  // [row][x][cin-chunk]
  const int b = blockIdx.z, y = blockIdx.y, x0 = blockIdx.x * 64;
  const int tid = threadIdx.x, wave = tid >> 5, lane = tid & 31;
  const int mtbase = (wave >> 2) * MPW;
  const int n0 = (wave & 3) * 16;
  const int n = lane & 15, khalf = lane >> 4;
  floatx8 acc[MPW] = {};

  for (int ch = 0; ch < NCH; ++ch) {
    __syncthreads();
    for (int i = tid; i < 3 * 66 * 4; i += 256) {
      int slot = i & 3; int t2 = i >> 2;
      int px = t2 % 66, ry = t2 / 66;
      int gy = y + ry - 1; gy = gy < 0 ? -gy : (gy >= HN ? 2 * HN - 2 - gy : gy);
      int gx = x0 + px - 1; gx = gx < 0 ? -gx : (gx >= WN ? 2 * WN - 2 - gx : gx);
      copy16_g2l(in + ((long)b * HWN + (long)gy * WN + gx) * CIN + ch * 32 + slot * 8,
                 &tin[ry][px][slot * 8]);
    }
    async_join();
    __syncthreads();
#pragma unroll
    for (int tap = 0; tap < 9; ++tap) {
      const int dy = tap / 3, dx = tap % 3;
      bf16x16 bb = frag_b16(&tin[dy][n0 + n + dx][khalf * 16]);
#pragma unroll
      for (int mi = 0; mi < MPW; ++mi) {
        int mtg = mtbase + mi;
        bf16x16 a = frag_b16(&wp[((((long)mtg * 9 + tap) * NCH + ch) * 32 + lane) * 16]);
        acc[mi] = wmma_bf16(a, bb, acc[mi]);
      }
    }
  }
#pragma unroll
  for (int mi = 0; mi < MPW; ++mi) {
    int mtg = mtbase + mi;
    float* dst = (mtg < 4) ? out0 : out1;
    int mt = mtg & 3;
#pragma unroll
    for (int r = 0; r < 8; ++r) {
      int m = mt * 16 + ((lane < 16) ? 0 : 8) + r;
      dst[((long)b * CN + m) * HWN + (long)y * WN + x0 + n0 + n] = acc[mi][r];
    }
  }
}

// ---------------------------------------------------------------------------
// BN training-mode stats -> affine coefs: y = x*coef[2c] + coef[2c+1]
// ---------------------------------------------------------------------------
__global__ void __launch_bounds__(256) k_stats(
    const float* __restrict__ t, const float* __restrict__ g,
    const float* __restrict__ be, float* __restrict__ coef) {
  const int c = blockIdx.x;
  float s = 0.f, q = 0.f;
  for (int b = 0; b < BN; ++b) {
    const float* p = t + ((long)b * CN + c) * HWN;
    for (int i = threadIdx.x; i < HWN; i += 256) { float v = p[i]; s += v; q += v * v; }
  }
  __shared__ float r1[256], r2[256];
  r1[threadIdx.x] = s; r2[threadIdx.x] = q; __syncthreads();
  for (int off = 128; off > 0; off >>= 1) {
    if (threadIdx.x < off) { r1[threadIdx.x] += r1[threadIdx.x + off];
                             r2[threadIdx.x] += r2[threadIdx.x + off]; }
    __syncthreads();
  }
  if (threadIdx.x == 0) {
    float N = (float)BN * HWN;
    float m = r1[0] / N;
    float var = r2[0] / N - m * m;
    float a = g[c] * rsqrtf(var + 1e-5f);
    coef[c * 2] = a; coef[c * 2 + 1] = be[c] - m * a;
  }
}

// BN+relu on t1 (fp32 channel-major) -> cb1 (bf16 pixel-major, conv2 input)
__global__ void k_bnrelu(const float* __restrict__ t1, const float* __restrict__ coef,
                         __bf16* __restrict__ cb1, long ntot) {
  for (long i = (long)blockIdx.x * blockDim.x + threadIdx.x; i < ntot;
       i += (long)gridDim.x * blockDim.x) {
    long p = i % HWN;
    int c = (int)((i / HWN) % CN);
    long b = i / ((long)HWN * CN);
    float v = t1[i] * coef[c * 2] + coef[c * 2 + 1];
    cb1[(b * HWN + p) * CN + c] = to_bf(v > 0.f ? v : 0.f);
  }
}

__global__ void k_final(const float* __restrict__ ti, const float* __restrict__ t2,
                        const float* __restrict__ cI, const float* __restrict__ c2,
                        float* __restrict__ out, long ntot) {
  for (long i = (long)blockIdx.x * blockDim.x + threadIdx.x; i < ntot;
       i += (long)gridDim.x * blockDim.x) {
    int c = (int)((i / HWN) % CN);
    float a = ti[i] * cI[c * 2] + cI[c * 2 + 1];
    a = a > 0.f ? a : 0.f;
    float v = a + (t2[i] * c2[c * 2] + c2[c * 2 + 1]);
    out[i] = v > 0.f ? v : 0.f;
  }
}

// ---------------------------------------------------------------------------
extern "C" void kernel_launch(void* const* d_in, const int* in_sizes, int n_in,
                              void* d_out, int out_size, void* d_ws, size_t ws_size,
                              hipStream_t stream) {
  const float* x_i  = (const float*)d_in[0];
  const float* x_e  = (const float*)d_in[1];
  const float* y    = (const float*)d_in[2];
  const float* temp = (const float*)d_in[3];
  const float* wq = (const float*)d_in[4];  const float* bq = (const float*)d_in[5];
  const float* wk = (const float*)d_in[6];  const float* bk = (const float*)d_in[7];
  const float* wv = (const float*)d_in[8];  const float* bv = (const float*)d_in[9];
  const float* wo = (const float*)d_in[10]; const float* bo = (const float*)d_in[11];
  const float* w_init = (const float*)d_in[12];
  const float* g_init = (const float*)d_in[14]; const float* be_init = (const float*)d_in[15];
  const float* w1 = (const float*)d_in[16];
  const float* g1 = (const float*)d_in[18]; const float* be1 = (const float*)d_in[19];
  const float* w2 = (const float*)d_in[20];
  const float* g2 = (const float*)d_in[22]; const float* be2 = (const float*)d_in[23];
  float* out = (float*)d_out;

  // ---- workspace layout ----
  float* W = (float*)d_ws;
  size_t o = 0;
  auto allocF = [&](size_t nf) { float* p = W + o; o += nf; return p; };
  auto allocB = [&](size_t nb) { __bf16* p = (__bf16*)(W + o); o += (nb + 1) / 2; return p; };
  const size_t T64 = (size_t)BN * CN * HWN;        // elements of one [B,64,HW] tensor
  __bf16* Qb   = allocB(T64);                      // channel-major bf16
  __bf16* Kxeb = allocB(T64);
  __bf16* Kxib = allocB(T64);
  __bf16* Vxib = allocB(T64);                      // pixel-major bf16 [b][hw][64]
  __bf16* Vxeb = allocB(T64);
  __bf16* fusedb = allocB((size_t)BN * HWN * 192); // pixel-major bf16 [b][hw][192]
  __bf16* cb1b = allocB(T64);                      // pixel-major bf16
  float* tI = allocF(T64);                         // conv outputs fp32 channel-major
  float* t1 = allocF(T64);
  float* t2 = allocF(T64);
  float* norms = allocF(3 * BN * CN);
  float* S     = allocF((size_t)BN * 2 * 8 * 64);
  float* Meff  = allocF((size_t)BN * 3 * 64 * 64);
  float* coefI = allocF(CN * 2);
  float* coef1 = allocF(CN * 2);
  float* coef2 = allocF(CN * 2);
  __bf16* wpq = allocB(4096);
  __bf16* wpk = allocB(4096);
  __bf16* wpv = allocB(4096);
  __bf16* mpk = allocB((size_t)BN * 3 * 4096);
  __bf16* wpc = allocB(2 * 110592);                // w_init || w1 packs
  __bf16* wp2 = allocB(36864);
  const long ntot = (long)T64;

  // 1) pre-swizzle static weights into WMMA A-fragment order
  k_pack_weights<<<64, 256, 0, stream>>>(wq, wpq, 64, 64, 1);
  k_pack_weights<<<64, 256, 0, stream>>>(wk, wpk, 64, 64, 1);
  k_pack_weights<<<64, 256, 0, stream>>>(wv, wpv, 64, 64, 1);
  k_pack_weights<<<256, 256, 0, stream>>>(w_init, wpc, 64, 192, 9);
  k_pack_weights<<<256, 256, 0, stream>>>(w1, wpc + 110592, 64, 192, 9);
  k_pack_weights<<<256, 256, 0, stream>>>(w2, wp2, 64, 64, 9);

  // 2) q/k/v projections
  const long BS = (long)CN * HWN;
  k_gemm64<float, false, false><<<dim3(HWN / 128, BN), 256, 0, stream>>>(
      y, BS, wpq, 0, bq, Qb, BS, 0, 0);
  k_gemm64<float, false, false><<<dim3(HWN / 128, BN), 256, 0, stream>>>(
      x_e, BS, wpk, 0, bk, Kxeb, BS, 0, 0);
  k_gemm64<float, false, false><<<dim3(HWN / 128, BN), 256, 0, stream>>>(
      x_i, BS, wpk, 0, bk, Kxib, BS, 0, 0);
  k_gemm64<float, false, true><<<dim3(HWN / 128, BN), 256, 0, stream>>>(
      x_i, BS, wpv, 0, bv, Vxib, BS, CN, 0);
  k_gemm64<float, false, true><<<dim3(HWN / 128, BN), 256, 0, stream>>>(
      x_e, BS, wpv, 0, bv, Vxeb, BS, CN, 0);

  // 3) norms + attention scores
  k_zero<<<8, 256, 0, stream>>>(S, (long)BN * 2 * 8 * 64);
  k_norms<<<dim3(CN, BN, 3), 256, 0, stream>>>(Qb, Kxeb, Kxib, norms);
  k_scores<<<dim3(4, 8, BN), 256, 0, stream>>>(Qb, Kxeb, Kxib, S);

  // 4) softmax + fold Wo -> Meff, pack per (b, slice)
  k_build_meff<<<1, 256, 0, stream>>>(S, norms, temp, wo, Meff);
  for (int b = 0; b < BN; ++b)
    for (int s = 0; s < 3; ++s)
      k_pack_weights<<<16, 256, 0, stream>>>(Meff + ((long)b * 3 + s) * 4096,
                                             mpk + ((long)b * 3 + s) * 4096, 64, 64, 1);

  // 5) fused[:, :, s*64..] = Meff_s @ V_src + bo   (pixel-major bf16 out)
  const __bf16* vsrc[3] = {Vxib, Vxeb, Vxib};
  for (int s = 0; s < 3; ++s)
    k_gemm64<__bf16, true, true><<<dim3(HWN / 128, BN), 256, 0, stream>>>(
        vsrc[s], BS, mpk + (long)s * 4096, (long)3 * 4096, bo,
        fusedb, (long)HWN * 192, 192, s * 64);

  // 6) dual 3x3 conv (w_init + w1 share one pass over `fused`)
  k_conv3x3<192, 8><<<dim3(WN / 64, HN, BN), 256, 0, stream>>>(fusedb, wpc, tI, t1);

  // 7) BN(t1)+relu -> cb1 (bf16 pixel-major), then conv(w2)
  k_stats<<<CN, 256, 0, stream>>>(t1, g1, be1, coef1);
  k_bnrelu<<<2048, 256, 0, stream>>>(t1, coef1, cb1b, ntot);
  k_conv3x3<64, 4><<<dim3(WN / 64, HN, BN), 256, 0, stream>>>(cb1b, wp2, t2, t2);

  // 8) stats for both branches, final residual + relu
  k_stats<<<CN, 256, 0, stream>>>(tI, g_init, be_init, coefI);
  k_stats<<<CN, 256, 0, stream>>>(t2, g2, be2, coef2);
  k_final<<<2048, 256, 0, stream>>>(tI, t2, coefI, coef2, out, ntot);
}
